// Nalui2Layer_47665547051318
// MI455X (gfx1250) — matrix operand order repaired and checked by reading, not compiled, not run on hardware
//
#include <hip/hip_runtime.h>
#include <hip/hip_bf16.h>
#include <math.h>

// ---------------------------------------------------------------------------
// NALU-i2 layer on MI455X (gfx1250, wave32, WMMA).
// Four fused f16 GEMMs (M=512, K=1024, N=512):
//   a1   = x    @ W1
//   mlog = logx @ W2
//   slog = neg  @ L      (L[i,o]  = log|1-2a|,  a = |W2| at remapped index)
//   par  = neg  @ P      (P[i,o]  = 1 if (1-2a)<0)  -> parity gives sign of ms1
// out = g*a1 + (1-g) * exp(min(mlog,20)) * clamp((-1)^par * exp(slog), -1, 1)
//
// K-loop is register double-buffered: iteration k+1's fragments are issued
// before iteration k's WMMAs so the s_wait before each WMMA group only drains
// the previous iteration's loads (no full s_wait_loadcnt 0 stall).
// ---------------------------------------------------------------------------

#define IN_DIM   1024
#define OUT_DIM  512
#define BATCH    512
#define SZ       (BATCH * IN_DIM)      // 524288 (== IN_DIM*OUT_DIM too)

typedef __attribute__((ext_vector_type(16))) _Float16 v16h;
typedef __attribute__((ext_vector_type(8)))  _Float16 v8h;
typedef __attribute__((ext_vector_type(8)))  float    v8f;

__device__ __forceinline__ float sigmoidf_(float v) {
    return 1.0f / (1.0f + __expf(-v));
}

// ---------------------------------------------------------------------------
// Prep 1: effective weights, pre-transposed to [o][i] (f16) so WMMA B-matrix
// fragments are contiguous per-lane loads.
// ---------------------------------------------------------------------------
__global__ void prep_weights_kernel(const float* __restrict__ w_hat1,
                                    const float* __restrict__ m_hat1,
                                    const float* __restrict__ w_hat2,
                                    const float* __restrict__ m_hat2,
                                    _Float16* __restrict__ W1t,
                                    _Float16* __restrict__ W2t,
                                    _Float16* __restrict__ Lt,
                                    _Float16* __restrict__ Pt) {
    int idx = blockIdx.x * blockDim.x + threadIdx.x;   // 0 .. SZ-1
    if (idx >= SZ) return;
    int o = idx >> 10;           // output index 0..511
    int i = idx & 1023;          // input  index 0..1023
    int src = i * OUT_DIM + o;   // row-major (in, out)

    float W1 = tanhf(w_hat1[src]) * sigmoidf_(m_hat1[src]);
    float W2 = tanhf(w_hat2[src]) * sigmoidf_(m_hat2[src]);

    // faithful tile/flatten pairing: a_{o,i} = |W2[2o + i/512, i % 512]|
    int r = 2 * o + (i >> 9);
    int c = i & 511;
    int src2 = r * OUT_DIM + c;
    float a = fabsf(tanhf(w_hat2[src2]) * sigmoidf_(m_hat2[src2]));
    float t = 1.0f - 2.0f * a;

    int dst = o * IN_DIM + i;    // transposed [o][i]
    W1t[dst] = (_Float16)W1;
    W2t[dst] = (_Float16)W2;
    Lt[dst]  = (_Float16)logf(fmaxf(fabsf(t), 1e-30f));
    Pt[dst]  = (_Float16)((t < 0.0f) ? 1.0f : 0.0f);
}

// ---------------------------------------------------------------------------
// Prep 2: three f16 A-matrix variants of x, row-major [b][i].
// ---------------------------------------------------------------------------
__global__ void prep_acts_kernel(const float* __restrict__ x,
                                 _Float16* __restrict__ X16,
                                 _Float16* __restrict__ LG16,
                                 _Float16* __restrict__ N16) {
    int idx = blockIdx.x * blockDim.x + threadIdx.x;   // 0 .. SZ-1
    if (idx >= SZ) return;
    float v = x[idx];
    X16[idx]  = (_Float16)v;
    LG16[idx] = (_Float16)logf(fmaxf(fabsf(v), 1e-7f));
    N16[idx]  = (_Float16)((v < 0.0f) ? 1.0f : 0.0f);
}

// ---------------------------------------------------------------------------
// Fragment loaders (wave32 WMMA); A and B share one contiguous k order, not the ISA table order.
// A (16x32 f16, row-major [m][k]): lane l -> row m0+(l&15),
//   halves 0..15 = K[kb + hi*16 .. +15]      (hi = l>>4)
// B (32x16 f16) from transposed weights [n][k]: lane l -> col n0+(l&15),
//   halves 0..15 = K[kb + hi*16 .. +15]  (contiguous 32B)
// ---------------------------------------------------------------------------
__device__ __forceinline__ v16h load_a_frag(const _Float16* __restrict__ base,
                                            int m, int kb, int hi) {
    const _Float16* p = base + (size_t)m * IN_DIM + kb + hi * 16;
    v8h lo = *(const v8h*)(p);
    v8h hh = *(const v8h*)(p + 8);
    return __builtin_shufflevector(lo, hh, 0, 1, 2, 3, 4, 5, 6, 7,
                                            8, 9, 10, 11, 12, 13, 14, 15);
}

__device__ __forceinline__ v16h load_b_frag(const _Float16* __restrict__ base,
                                            int n, int kb, int hi) {
    return *(const v16h*)(base + (size_t)n * IN_DIM + kb + hi * 16);
}

__device__ __forceinline__ v8f wmma_f16(v16h a, v16h b, v8f c) {
    return __builtin_amdgcn_wmma_f32_16x16x32_f16(
        /*neg_a=*/false, a, /*neg_b=*/false, b,
        /*c_mod=*/(short)0, c, /*reuse_a=*/false, /*reuse_b=*/false);
}

struct Frags {
    v16h ax, al, an;          // A variants: x, log|x|, 1[x<0]
    v16h b1, b2, bl, bp;      // B variants: W1t, W2t, Lt, Pt
};

// ---------------------------------------------------------------------------
// Fused 4-GEMM + epilogue. Block = 256 threads = 8 waves; waves tiled 4(M)x2(N),
// each wave owns one 16x16 output tile across all four accumulators.
// Grid: x = N/32 = 16, y = M/64 = 8.
// ---------------------------------------------------------------------------
__global__ __launch_bounds__(256)
void nalu_wmma_kernel(const _Float16* __restrict__ X16,
                      const _Float16* __restrict__ LG16,
                      const _Float16* __restrict__ N16,
                      const _Float16* __restrict__ W1t,
                      const _Float16* __restrict__ W2t,
                      const _Float16* __restrict__ Lt,
                      const _Float16* __restrict__ Pt,
                      const float* __restrict__ G1,
                      float* __restrict__ out) {
    const int tid  = threadIdx.x;
    const int lane = tid & 31;
    const int w    = tid >> 5;                 // wave id 0..7
    const int m0   = blockIdx.y * 64 + (w >> 1) * 16;
    const int n0   = blockIdx.x * 32 + (w & 1) * 16;
    const int lm   = lane & 15;
    const int hi   = lane >> 4;
    const int am   = m0 + lm;                  // A row for this lane
    const int bn   = n0 + lm;                  // B col for this lane

    v8f acc_a = {}, acc_m = {}, acc_s = {}, acc_p = {};

    auto load_all = [&](int kb) {
        Frags f;
        f.ax = load_a_frag(X16,  am, kb, hi);
        f.al = load_a_frag(LG16, am, kb, hi);
        f.an = load_a_frag(N16,  am, kb, hi);
        f.b1 = load_b_frag(W1t, bn, kb, hi);
        f.b2 = load_b_frag(W2t, bn, kb, hi);
        f.bl = load_b_frag(Lt,  bn, kb, hi);
        f.bp = load_b_frag(Pt,  bn, kb, hi);
        return f;
    };
    auto accumulate = [&](const Frags& f) {
        acc_a = wmma_f16(f.ax, f.b1, acc_a);
        acc_m = wmma_f16(f.al, f.b2, acc_m);
        acc_s = wmma_f16(f.an, f.bl, acc_s);
        acc_p = wmma_f16(f.an, f.bp, acc_p);
    };

    // Software-pipelined K loop: prefetch next step's 7 fragments before
    // consuming the current ones, last step peeled.
    Frags cur = load_all(0);
    #pragma unroll 2
    for (int kb = 0; kb < IN_DIM - 32; kb += 32) {
        Frags nxt = load_all(kb + 32);
        accumulate(cur);
        cur = nxt;
    }
    accumulate(cur);

    // Epilogue. C/D layout: VGPR e -> M = e + hi*8, N = lane&15 (ISA 7.12.2).
    const int col = n0 + lm;
    const float g = sigmoidf_(G1[col]);
    #pragma unroll
    for (int e = 0; e < 8; ++e) {
        const int row = m0 + hi * 8 + e;
        float a1 = acc_a[e];
        float m1 = __expf(fminf(acc_m[e], 20.0f));
        int  par = __float2int_rn(acc_p[e]) & 1;      // exact: integer-valued f32
        float ms = __expf(acc_s[e]);
        ms = par ? -ms : ms;
        ms = fminf(fmaxf(ms, -1.0f), 1.0f);           // clip(ms1, -1, 1)
        out[(size_t)row * OUT_DIM + col] = g * a1 + (1.0f - g) * m1 * ms;
    }
}

// ---------------------------------------------------------------------------
// Launcher. Inputs: x, w_hat1, m_hat1, w_hat2, m_hat2, G1 (all f32).
// Workspace layout (f16): [W1t | W2t | Lt | Pt | X16 | LG16 | N16], 7 MB.
// ---------------------------------------------------------------------------
extern "C" void kernel_launch(void* const* d_in, const int* in_sizes, int n_in,
                              void* d_out, int out_size, void* d_ws, size_t ws_size,
                              hipStream_t stream) {
    const float* x      = (const float*)d_in[0];
    const float* w_hat1 = (const float*)d_in[1];
    const float* m_hat1 = (const float*)d_in[2];
    const float* w_hat2 = (const float*)d_in[3];
    const float* m_hat2 = (const float*)d_in[4];
    const float* G1     = (const float*)d_in[5];
    float* out = (float*)d_out;

    _Float16* ws  = (_Float16*)d_ws;
    _Float16* W1t = ws + 0 * (size_t)SZ;
    _Float16* W2t = ws + 1 * (size_t)SZ;
    _Float16* Lt  = ws + 2 * (size_t)SZ;
    _Float16* Pt  = ws + 3 * (size_t)SZ;
    _Float16* X16 = ws + 4 * (size_t)SZ;
    _Float16* LG  = ws + 5 * (size_t)SZ;
    _Float16* N16 = ws + 6 * (size_t)SZ;

    const int threads = 256;
    const int blocks  = SZ / threads;   // 2048

    prep_weights_kernel<<<blocks, threads, 0, stream>>>(
        w_hat1, m_hat1, w_hat2, m_hat2, W1t, W2t, Lt, Pt);
    prep_acts_kernel<<<blocks, threads, 0, stream>>>(x, X16, LG, N16);

    dim3 grid(OUT_DIM / 32, BATCH / 64);   // 16 x 8
    nalu_wmma_kernel<<<grid, 256, 0, stream>>>(
        X16, LG, N16, W1t, W2t, Lt, Pt, G1, out);
}
